// PLCCLoss_26680336843078
// MI455X (gfx1250) — matrix operand order repaired and checked by reading, not compiled, not run on hardware
//
#include <hip/hip_runtime.h>

// PLCC loss (InfoNCE + continuity), N=4096, D=64.
// sim = fn @ fn.T via v_wmma_f32_16x16x32_f16 (K=64 -> 2 chained WMMAs per
// 16x16 tile), epilogue fully fused. All data L2-resident (~1.5 MB), so the
// kernel is compute bound; B-strip is staged once per block into LDS by the
// Tensor Data Mover (tensor_load_to_lds + s_wait_tensorcnt) with pad_enable
// so WMMA B-fragment ds_load_b128 reads are bank-conflict free.

#define NPTS 4096
#define DDIM 64
#define GAMMA 0.5f
#define RADIUS 1.0f
#define SELF_EPS 1e-6f
#define COS_EPS 1e-8f
#define NCE_EPS 1e-6f
#define EXP2_SCALE 14.426950408889634f   // 10 / ln(2): exp(sim/0.1)=2^(sim*this)

#define COLS_PER_BLOCK 256
#define LDS_ROW_H 72                     // 64 data halves + 8 pad halves (TDM pad)

typedef _Float16 h8   __attribute__((ext_vector_type(8)));
typedef _Float16 v16h __attribute__((ext_vector_type(16)));
typedef float    v8f  __attribute__((ext_vector_type(8)));
typedef unsigned int u32x4 __attribute__((ext_vector_type(4)));
typedef unsigned int u32x8 __attribute__((ext_vector_type(8)));

#if __has_builtin(__builtin_amdgcn_sqrtf)
#define FAST_SQRT(x) __builtin_amdgcn_sqrtf(x)
#else
#define FAST_SQRT(x) sqrtf(x)
#endif
#if __has_builtin(__builtin_amdgcn_exp2f)
#define FAST_EXP2(x) __builtin_amdgcn_exp2f(x)
#else
#define FAST_EXP2(x) exp2f(x)
#endif

// ---------------------------------------------------------------- prep ----
__global__ __launch_bounds__(256) void plcc_prep(const float* __restrict__ feat,
                                                 _Float16* __restrict__ fnh) {
  int row = blockIdx.x * blockDim.x + threadIdx.x;
  if (row >= NPTS) return;
  const float* f = feat + row * DDIM;
  float s = 0.0f;
#pragma unroll
  for (int k = 0; k < DDIM; ++k) s += f[k] * f[k];
  float inv = 1.0f / fmaxf(FAST_SQRT(s), COS_EPS);
  _Float16* o = fnh + row * DDIM;
#pragma unroll
  for (int k = 0; k < DDIM; ++k) o[k] = (_Float16)(f[k] * inv);
}

// ---------------------------------------------------------------- zero ----
__global__ __launch_bounds__(256) void plcc_zero(float* __restrict__ pos_sum,
                                                 float* __restrict__ neg_sum,
                                                 float* __restrict__ cont_sum) {
  int i = blockIdx.x * blockDim.x + threadIdx.x;
  if (i < NPTS) { pos_sum[i] = 0.0f; neg_sum[i] = 0.0f; }
  if (i == 0) cont_sum[0] = 0.0f;
}

// ---------------------------------------------------------------- tile ----
// Grid: x = 16 column blocks of 256 cols, y = 32 row blocks of 128 rows.
// 8 waves/block; wave w owns rows [row0, row0+16) and sweeps 16 col tiles.
__global__ __launch_bounds__(256) void plcc_tile(const _Float16* __restrict__ fnh,
                                                 const float* __restrict__ coords,
                                                 const int*   __restrict__ labels,
                                                 float* __restrict__ pos_sum,
                                                 float* __restrict__ neg_sum,
                                                 float* __restrict__ cont_sum) {
  __shared__ __align__(16) _Float16 sB[COLS_PER_BLOCK * LDS_ROW_H]; // 36 KB
  __shared__ float sCoord[COLS_PER_BLOCK * 3];                      // 3 KB
  __shared__ float sWc[COLS_PER_BLOCK];                             // 1 KB

  const int tid  = threadIdx.x;
  const int lane = tid & 31;
  const int wave = tid >> 5;
  const int hgrp = lane >> 4;     // 0: lanes 0-15, 1: lanes 16-31
  const int m    = lane & 15;
  const int row0 = blockIdx.y * 128 + wave * 16;
  const int colbase = blockIdx.x * COLS_PER_BLOCK;

  // ---- TDM: stage 256 cols x 64 halves of fn into LDS, once per block.
  // 2-D descriptor (groups 0+1): data_size=2B, tile 64 x 256, stride 64,
  // pad_enable: +4 DWORDs per 32-DWORD row -> LDS row stride 144 B, which
  // makes the 16-lane B-fragment reads a conflict-free bank permutation.
  if (wave == 0) {
    unsigned long long gaddr =
        (unsigned long long)(const void*)(fnh + (size_t)colbase * DDIM);
    unsigned ldsoff = (unsigned)(unsigned long long)(const void*)&sB[0];
    u32x4 g0 = { 1u,                                   // count=1, user mode
                 ldsoff,                               // lds_addr
                 (unsigned)gaddr,                      // global_addr[31:0]
                 (unsigned)(gaddr >> 32) | (2u << 30) };// addr[56:32] | type=2
    u32x8 g1 = { (1u << 16) | (1u << 20) | (4u << 22) | (3u << 25),
                 // data_size=2B | pad_enable | pad_interval=32DW | pad=4DW
                 (unsigned)DDIM << 16,                 // tensor_dim0 = 64
                 (unsigned)NPTS << 16,                 // tensor_dim1 = 4096
                 (unsigned)DDIM << 16,                 // tile_dim0 = 64
                 (unsigned)COLS_PER_BLOCK,             // tile_dim1 = 256
                 (unsigned)DDIM,                       // tensor_dim0_stride=64
                 0u, 0u };
    asm volatile("tensor_load_to_lds %0, %1" :: "s"(g0), "s"(g1) : "memory");
    __builtin_amdgcn_s_wait_tensorcnt(0);
  }
  // ---- stage column coords / label weights (all 256 threads).
  {
    int c = colbase + tid;
    sCoord[tid * 3 + 0] = coords[c * 3 + 0];
    sCoord[tid * 3 + 1] = coords[c * 3 + 1];
    sCoord[tid * 3 + 2] = coords[c * 3 + 2];
    sWc[tid] = (labels[c] == 2) ? 1.0f : 0.0f;
  }
  __syncthreads();

  // ---- A fragments: 16x32 f16 layout (ISA 7.12.2), loaded once per wave.
  v16h a0, a1;
  {
    const _Float16* rp = fnh + (size_t)(row0 + m) * DDIM;
    h8 lo0 = *(const h8*)(rp +      hgrp * 8);
    h8 hi0 = *(const h8*)(rp + 16 + hgrp * 8);
    h8 lo1 = *(const h8*)(rp + 32 + hgrp * 8);
    h8 hi1 = *(const h8*)(rp + 48 + hgrp * 8);
#pragma unroll
    for (int i = 0; i < 8; ++i) {
      a0[i] = lo0[i]; a0[i + 8] = hi0[i];
      a1[i] = lo1[i]; a1[i + 8] = hi1[i];
    }
  }

  // ---- Row-side coords / label weights (C layout: M = r + 8*hgrp).
  float rcx[8], rcy[8], rcz[8], wr[8];
#pragma unroll
  for (int r = 0; r < 8; ++r) {
    int row = row0 + (hgrp << 3) + r;
    rcx[r] = coords[row * 3 + 0];
    rcy[r] = coords[row * 3 + 1];
    rcz[r] = coords[row * 3 + 2];
    wr[r]  = (labels[row] == 2) ? 1.0f : 0.0f;
  }

  float ps[8], ns[8];
#pragma unroll
  for (int r = 0; r < 8; ++r) { ps[r] = 0.0f; ns[r] = 0.0f; }
  float cont = 0.0f;

  for (int ct = 0; ct < 16; ++ct) {
    const int nloc = ct * 16 + m;

    // ---- B fragments from LDS: 32x16 f16 layout; lane group h, element e:
    // k = 8h + e (e<8), 16 + 8h + (e-8) (e>=8) (+ kstep). LDS row stride = 72 halves (TDM-padded).
    v16h b0, b1;
    {
      const _Float16* cp = &sB[nloc * LDS_ROW_H + hgrp * 8];
      h8 l0 = *(const h8*)(cp);
      h8 h0 = *(const h8*)(cp + 16);
      h8 l1 = *(const h8*)(cp + 32);
      h8 h1 = *(const h8*)(cp + 48);
#pragma unroll
      for (int i = 0; i < 8; ++i) {
        b0[i] = l0[i]; b0[i + 8] = h0[i];
        b1[i] = l1[i]; b1[i + 8] = h1[i];
      }
    }

    v8f c = {};
    c = __builtin_amdgcn_wmma_f32_16x16x32_f16(false, a0, false, b0,
                                               (short)0, c, false, false);
    c = __builtin_amdgcn_wmma_f32_16x16x32_f16(false, a1, false, b1,
                                               (short)0, c, false, false);

    const float ccx = sCoord[nloc * 3 + 0];
    const float ccy = sCoord[nloc * 3 + 1];
    const float ccz = sCoord[nloc * 3 + 2];
    const float wc  = sWc[nloc];

#pragma unroll
    for (int r = 0; r < 8; ++r) {
      float sim = c[r];
      float dx = rcx[r] - ccx, dy = rcy[r] - ccy, dz = rcz[r] - ccz;
      float d2 = dx * dx + dy * dy + dz * dz;
      // mask on d^2 (monotonic): d<1 <=> d2<1 ; d>1e-6 <=> d2>1e-12
      bool  pos = (d2 < RADIUS * RADIUS) && (d2 > SELF_EPS * SELF_EPS);
      float pw  = wr[r] * wc;
      float e   = FAST_EXP2(sim * EXP2_SCALE) * pw;   // exp(sim/0.1)*pair_w
      ps[r] += pos ? e : 0.0f;
      ns[r] += pos ? 0.0f : e;
      float d = FAST_SQRT(d2);                        // only feeds continuity
      cont  += pos ? pw * fabsf((1.0f - sim) - d) : 0.0f;
    }
  }

  // ---- Row sums: reduce across the 16-lane half-group, one atomic/row.
#pragma unroll
  for (int r = 0; r < 8; ++r) {
    float p = ps[r], q = ns[r];
#pragma unroll
    for (int off = 1; off < 16; off <<= 1) {
      p += __shfl_xor(p, off, 32);
      q += __shfl_xor(q, off, 32);
    }
    if (m == 0) {
      int row = row0 + (hgrp << 3) + r;
      atomicAdd(&pos_sum[row], p);
      atomicAdd(&neg_sum[row], q);
    }
  }
  // ---- Continuity: full-wave reduce, one atomic per wave.
#pragma unroll
  for (int off = 1; off < 32; off <<= 1) cont += __shfl_xor(cont, off, 32);
  if (lane == 0) atomicAdd(cont_sum, cont);
}

// ------------------------------------------------------------ finalize ----
__global__ __launch_bounds__(256) void plcc_finalize(const float* __restrict__ pos_sum,
                                                     const float* __restrict__ neg_sum,
                                                     const float* __restrict__ cont_sum,
                                                     const int*   __restrict__ labels,
                                                     float* __restrict__ out) {
  __shared__ float s_info[256];
  __shared__ float s_m[256];
  int tid = threadIdx.x;
  float info = 0.0f, mcnt = 0.0f;
  for (int i = tid; i < NPTS; i += 256) {
    if (labels[i] == 2) {
      mcnt += 1.0f;
      float p = pos_sum[i];
      float q = neg_sum[i];
      info -= __logf(p / (q + p + NCE_EPS));
    }
  }
  s_info[tid] = info;
  s_m[tid]    = mcnt;
  __syncthreads();
  for (int s = 128; s > 0; s >>= 1) {
    if (tid < s) { s_info[tid] += s_info[tid + s]; s_m[tid] += s_m[tid + s]; }
    __syncthreads();
  }
  if (tid == 0) {
    float M = s_m[0];
    out[0] = s_info[0] / M + GAMMA * (cont_sum[0] / (M * M));
  }
}

// -------------------------------------------------------------- launch ----
extern "C" void kernel_launch(void* const* d_in, const int* in_sizes, int n_in,
                              void* d_out, int out_size, void* d_ws, size_t ws_size,
                              hipStream_t stream) {
  (void)in_sizes; (void)n_in; (void)out_size; (void)ws_size;
  const float* features = (const float*)d_in[0];   // [4096,64] f32
  const int*   labels   = (const int*)  d_in[1];   // [4096]    i32
  const float* coords   = (const float*)d_in[2];   // [4096,3]  f32
  float*       out      = (float*)d_out;           // scalar f32

  char* ws = (char*)d_ws;
  _Float16* fnh   = (_Float16*)ws;                              // 512 KB
  float* pos_sum  = (float*)(ws + (size_t)NPTS * DDIM * 2);     // 16 KB
  float* neg_sum  = pos_sum + NPTS;                             // 16 KB
  float* cont_sum = neg_sum + NPTS;                             // 4 B

  plcc_prep<<<NPTS / 256, 256, 0, stream>>>(features, fnh);
  plcc_zero<<<NPTS / 256, 256, 0, stream>>>(pos_sum, neg_sum, cont_sum);

  dim3 grid(NPTS / COLS_PER_BLOCK, NPTS / 128);   // (16, 32)
  plcc_tile<<<grid, 256, 0, stream>>>(fnh, coords, labels,
                                      pos_sum, neg_sum, cont_sum);

  plcc_finalize<<<1, 256, 0, stream>>>(pos_sum, neg_sum, cont_sum, labels, out);
}
